// RN_Module_77979426226382
// MI455X (gfx1250) — matrix operand order repaired and checked by reading, not compiled, not run on hardware
//
#include <hip/hip_runtime.h>
#include <hip/hip_bf16.h>

typedef __attribute__((ext_vector_type(16))) _Float16 v16h;
typedef __attribute__((ext_vector_type(8)))  _Float16 v8h;
typedef __attribute__((ext_vector_type(8)))  float    v8f;

#define WMMA_F16(acc, a, b) \
    __builtin_amdgcn_wmma_f32_16x16x32_f16(false, (a), false, (b), (short)0, (acc), false, false)

// ---------------------------------------------------------------------------
// Pack kernels: f32 row-major W[K][N] -> f16 transposed Wt[N][K]
// ---------------------------------------------------------------------------
__global__ void pack_wt(const float* __restrict__ W, _Float16* __restrict__ Wt,
                        int K, int N) {
    int idx = blockIdx.x * blockDim.x + threadIdx.x;
    if (idx < K * N) {
        int n = idx / K, k = idx % K;
        Wt[idx] = (_Float16)W[k * N + n];
    }
}

// W0 is [256][256] (rows 0..127 multiply state[i], rows 128..255 multiply state[j]).
// Build W0t f16 [512][128]:  n<256 -> column n of top half, n>=256 -> column n-256 of bottom half.
__global__ void pack_w0(const float* __restrict__ W0, _Float16* __restrict__ Wt) {
    int idx = blockIdx.x * blockDim.x + threadIdx.x;
    if (idx < 512 * 128) {
        int n = idx / 128, k = idx % 128;
        float v = (n < 256) ? W0[k * 256 + n] : W0[(128 + k) * 256 + (n - 256)];
        Wt[idx] = (_Float16)v;
    }
}

// ---------------------------------------------------------------------------
// Generic WMMA GEMM: Out[M,N] = act(A[M,K] @ Wt^T + bias), one wave per 16x16 tile
// A: f32 row-major. Wt: f16 [N][K] (transposed). bias may be null. relu optional.
// ---------------------------------------------------------------------------
__global__ void gemm_bias_relu(const float* __restrict__ A,
                               const _Float16* __restrict__ Wt,
                               const float* __restrict__ bias,
                               float* __restrict__ Out,
                               int M, int N, int K, int relu) {
    int wid    = (blockIdx.x * blockDim.x + threadIdx.x) >> 5;
    int nwaves = (gridDim.x * blockDim.x) >> 5;
    int lane   = threadIdx.x & 31;
    int ln15   = lane & 15;
    int hf     = lane >> 4;

    int nT = N >> 4, mT = M >> 4;
    int tiles = mT * nT;
    int ksteps = K >> 5;

    for (int t = wid; t < tiles; t += nwaves) {
        int mt = t / nT, nt = t % nT;
        const float*    Arow = A  + (size_t)(mt * 16 + ln15) * K;
        const _Float16* Brow = Wt + (size_t)(nt * 16 + ln15) * K + hf * 16;

        v8f acc = {};
        for (int ks = 0; ks < ksteps; ++ks) {
            // A fragment: one contiguous 16-float run per lane (k = 16*hf + e)
            const float* p0 = Arow + ks * 32 + hf * 16;
            const float* p1 = p0 + 8;
            v16h a;
#pragma unroll
            for (int e = 0; e < 8; ++e) {
                a[e]     = (_Float16)p0[e];
                a[e + 8] = (_Float16)p1[e];
            }
            v16h b = *(const v16h*)(Brow + ks * 32);
            acc = WMMA_F16(acc, a, b);
        }

        float bv = bias ? bias[nt * 16 + ln15] : 0.0f;
        float* Ocol = Out + (size_t)(nt * 16 + ln15);
#pragma unroll
        for (int v = 0; v < 8; ++v) {
            int row = mt * 16 + v + hf * 8;
            float x = acc[v] + bv;
            if (relu) x = fmaxf(x, 0.0f);
            Ocol[(size_t)row * N] = x;
        }
    }
}

// ---------------------------------------------------------------------------
// Pair kernel: one workgroup per batch b (480 threads = 15 waves, wave w = M-tile w).
// gW1 (f16, 128KB) staged into LDS with CDNA5 async copies (ASYNCcnt path),
// U/V staged as f16 (16KB). A-fragments built with packed f16 math:
// a = relu(U[i]+V[j]+gb0); h2 = relu(a @ gW1 + gb1); pooled[b] = sum_p h2.
// ---------------------------------------------------------------------------
__global__ __launch_bounds__(480)
void pair_gemm_pool(const float* __restrict__ UV,       // [B*16][512]: U cols 0..255, V cols 256..511
                    const _Float16* __restrict__ W1t,   // [256][256] transposed f16
                    const float* __restrict__ gb0,      // [256]
                    const float* __restrict__ gb1,      // [256]
                    float* __restrict__ pooled) {       // [B][256]
    __shared__ _Float16 lds_W1[256 * 256];              // 131072 B
    __shared__ _Float16 lds_U[16][256];                 //   8192 B
    __shared__ _Float16 lds_V[16][256];                 //   8192 B
    __shared__ _Float16 lds_b0[256];                    //    512 B (f16, joins packed math)
    __shared__ float    lds_b1[256];                    //   1024 B
    __shared__ float    lds_pool[256];                  //   1024 B

    const int b   = blockIdx.x;
    const int tid = threadIdx.x;

    // --- Async-copy gW1 (f16, 8192 x 16B chunks) global -> LDS (ASYNCcnt path)
    {
        const char*    gsrc    = (const char*)W1t;
        const unsigned ldsBase = (unsigned)(uintptr_t)(void*)lds_W1;
        for (int idx = tid; idx < 8192; idx += 480) {
            unsigned    loff = ldsBase + (unsigned)idx * 16u;
            const void* gptr = gsrc + (size_t)idx * 16u;
            asm volatile("global_load_async_to_lds_b128 %0, %1, off"
                         :: "v"(loff), "v"(gptr) : "memory");
        }
    }

    // --- Stage U[b], V[b] (16x512 f32 -> f16) and biases into LDS
    const float* src = UV + (size_t)b * 16 * 512;
    for (int idx = tid; idx < 16 * 512; idx += 480) {
        int s = idx >> 9, c = idx & 511;
        float v = src[idx];
        if (c < 256) lds_U[s][c]       = (_Float16)v;
        else         lds_V[s][c - 256] = (_Float16)v;
    }
    for (int c = tid; c < 256; c += 480) {
        lds_b0[c]   = (_Float16)gb0[c];
        lds_b1[c]   = gb1[c];
        lds_pool[c] = 0.0f;
    }
    asm volatile("s_wait_asynccnt 0x0" ::: "memory");   // my async copies landed
    __syncthreads();                                    // everyone's copies visible

    const int w    = tid >> 5;   // wave id 0..14  -> M-tile (16 pairs each, 15*16 = 240 pairs)
    const int lane = tid & 31;
    const int ln15 = lane & 15;
    const int hf   = lane >> 4;

    // A row for this lane: pair p = w*16 + ln15 in itertools.permutations order
    const int p = w * 16 + ln15;
    const int i = p / 15;
    const int r = p % 15;
    const int j = r + (r >= i ? 1 : 0);
    const _Float16* Urow = &lds_U[i][0];
    const _Float16* Vrow = &lds_V[j][0];

    // Build all 8 K-step A fragments once, entirely in packed f16
    // (fragment = one contiguous 16-element run per lane: k0..k0+7, k0+8..k0+15)
    v16h afrag[8];
#pragma unroll
    for (int ks = 0; ks < 8; ++ks) {
        const int k0 = ks * 32 + hf * 16;
        const int k1 = k0 + 8;
        v8h lo = *(const v8h*)(Urow + k0) + *(const v8h*)(Vrow + k0)
               + *(const v8h*)(&lds_b0[k0]);
        v8h hi = *(const v8h*)(Urow + k1) + *(const v8h*)(Vrow + k1)
               + *(const v8h*)(&lds_b0[k1]);
        v8h z = {};
        lo = __builtin_elementwise_max(lo, z);          // v_pk_max f16
        hi = __builtin_elementwise_max(hi, z);
        afrag[ks] = __builtin_shufflevector(lo, hi, 0, 1, 2, 3, 4, 5, 6, 7,
                                            8, 9, 10, 11, 12, 13, 14, 15);
    }

    // Sweep all 16 N-tiles of gW1 (B-fragments from LDS: ds_load_b128 pairs)
    for (int nt = 0; nt < 16; ++nt) {
        const _Float16* Brow = &lds_W1[(nt * 16 + ln15) * 256 + hf * 16];
        v8f acc = {};
#pragma unroll
        for (int ks = 0; ks < 8; ++ks) {
            v16h bf = *(const v16h*)(Brow + ks * 32);
            acc = WMMA_F16(acc, afrag[ks], bf);
        }
        const float bv = lds_b1[nt * 16 + ln15];
        float s = 0.0f;
#pragma unroll
        for (int v = 0; v < 8; ++v) s += fmaxf(acc[v] + bv, 0.0f);
        atomicAdd(&lds_pool[nt * 16 + ln15], s);   // ds_add_f32, lanes L / L+16 collide safely
    }
    __syncthreads();

    for (int c = tid; c < 256; c += 480) pooled[(size_t)b * 256 + c] = lds_pool[c];
}

// ---------------------------------------------------------------------------
// Launch
// ---------------------------------------------------------------------------
extern "C" void kernel_launch(void* const* d_in, const int* in_sizes, int n_in,
                              void* d_out, int out_size, void* d_ws, size_t ws_size,
                              hipStream_t stream) {
    const float* state = (const float*)d_in[0];  // [1024,16,128]
    const float* gW0   = (const float*)d_in[1];  // [256,256]
    const float* gb0   = (const float*)d_in[2];  // [256]
    const float* gW1   = (const float*)d_in[3];  // [256,256]
    const float* gb1   = (const float*)d_in[4];  // [256]
    const float* fW0   = (const float*)d_in[5];  // [256,256]
    const float* fb0   = (const float*)d_in[6];  // [256]
    const float* fW1   = (const float*)d_in[7];  // [256,256]
    const float* fb1   = (const float*)d_in[8];  // [256]
    float* out = (float*)d_out;                  // [1024,256]

    // Workspace layout (all offsets 32B aligned)
    char* ws = (char*)d_ws;
    float*    UV     = (float*)(ws);                          // 16384*512*4 = 33,554,432
    float*    pooled = (float*)(ws + 33554432);               // 1024*256*4  =  1,048,576
    float*    tmp    = (float*)(ws + 34603008);               // 1024*256*4  =  1,048,576
    _Float16* W0t    = (_Float16*)(ws + 35651584);            // 512*128*2   =    131,072
    _Float16* W1t    = (_Float16*)(ws + 35782656);            // 256*256*2
    _Float16* F0t    = (_Float16*)(ws + 35913728);            // 256*256*2
    _Float16* F1t    = (_Float16*)(ws + 36044800);            // 256*256*2

    // 1) Pack weights to transposed f16
    pack_w0<<<(512 * 128 + 255) / 256, 256, 0, stream>>>(gW0, W0t);
    pack_wt<<<(256 * 256 + 255) / 256, 256, 0, stream>>>(gW1, W1t, 256, 256);
    pack_wt<<<(256 * 256 + 255) / 256, 256, 0, stream>>>(fW0, F0t, 256, 256);
    pack_wt<<<(256 * 256 + 255) / 256, 256, 0, stream>>>(fW1, F1t, 256, 256);

    // 2) UV = state[16384,128] @ W0cat[128,512]   (no bias/relu; gb0 applied in pair kernel)
    gemm_bias_relu<<<1024, 256, 0, stream>>>(state, W0t, nullptr, UV,
                                             16384, 512, 128, 0);

    // 3) Pair GEMM + pooled sum, one workgroup per batch
    pair_gemm_pool<<<1024, 480, 0, stream>>>(UV, W1t, gb0, gb1, pooled);

    // 4) f-MLP: two WMMA layers with bias + relu
    gemm_bias_relu<<<128, 256, 0, stream>>>(pooled, F0t, fb0, tmp, 1024, 256, 256, 1);
    gemm_bias_relu<<<128, 256, 0, stream>>>(tmp,    F1t, fb1, out, 1024, 256, 256, 1);
}
